// GraphConvolution_71811853189638
// MI455X (gfx1250) — matrix operand order repaired and checked by reading, not compiled, not run on hardware
//
#include <hip/hip_runtime.h>

typedef __attribute__((ext_vector_type(16))) __bf16 v16bf;
typedef __attribute__((ext_vector_type(8)))  float  v8f;
typedef __attribute__((ext_vector_type(4)))  unsigned int u32x4;
typedef __attribute__((ext_vector_type(8)))  int i32x8;
typedef __attribute__((ext_vector_type(4)))  int i32x4;

union FragBF { v16bf v; __bf16 e[16]; uint4 q[2]; };

#define M_TILE 128
#define N_TILE 128
#define K_TILE 32
#define KPAD_F32  36   // fp32 rows: 32 data + 4 pad dwords = 144B, 16B-aligned
#define KPAD_BF16 40   // bf16 rows: 32 data (64B) + 4 pad dwords = 80B, 16B-aligned

#define HAVE_TDM __has_builtin(__builtin_amdgcn_tensor_load_to_lds)

#if HAVE_TDM
#if __has_include(<hip/amd_detail/amd_gfx1250_TDM.h>)
#define TDM_CALL(g0, g1, g2, g3) \
    __builtin_amdgcn_tensor_load_to_lds(g0, g1, g2, g3, (i32x8)0, 0)
#else
#define TDM_CALL(g0, g1, g2, g3) \
    __builtin_amdgcn_tensor_load_to_lds(g0, g1, g2, g3, 0)
#endif

// D# group0: count=1, lds byte addr, 57-bit global byte addr, type=2 ("image")
__device__ __forceinline__ u32x4 tdm_g0(unsigned lds_addr, unsigned long long gaddr) {
    u32x4 g;
    g[0] = 1u;
    g[1] = lds_addr;
    g[2] = (unsigned)gaddr;
    g[3] = (unsigned)((gaddr >> 32) & 0x01FFFFFFu) | (2u << 30);
    return g;
}

// D# group1: data_size enc (0=1B,1=2B,2=4B), pad_interval/amount encodings per ISA 8.4
__device__ __forceinline__ i32x8 tdm_g1(unsigned dsize, unsigned pi, unsigned pa,
                                        unsigned td0, unsigned td1,
                                        unsigned tile0, unsigned tile1, unsigned tile2,
                                        unsigned long long s0, unsigned long long s1) {
    const unsigned dw0 = (dsize << 16) | (1u << 20) | (pi << 22) | (pa << 25);
    i32x8 g;
    g[0] = (int)dw0;                                       // wg_mask=0 (no cluster)
    g[1] = (int)((td0 & 0xFFFFu) << 16);                   // tensor_dim0[15:0]
    g[2] = (int)((td0 >> 16) | ((td1 & 0xFFFFu) << 16));   // td0[31:16] | td1[15:0]
    g[3] = (int)((td1 >> 16) | (tile0 << 16));             // td1[31:16] | tile_dim0
    g[4] = (int)(tile1 | (tile2 << 16));                   // tile_dim1 | tile_dim2
    g[5] = (int)(unsigned)s0;                              // dim0_stride[31:0]
    g[6] = (int)(((unsigned)(s0 >> 32) & 0xFFFFu) |
                 (((unsigned)s1 & 0xFFFFu) << 16));        // s0[47:32] | s1[15:0]
    g[7] = (int)(unsigned)(s1 >> 16);                      // s1[47:16]
    return g;
}

__device__ __forceinline__ void tdm_issue(unsigned lds_addr, const void* gptr,
                                          unsigned dsize, unsigned pi, unsigned pa,
                                          unsigned td0, unsigned td1, unsigned td2,
                                          unsigned tile0, unsigned tile1, unsigned tile2,
                                          unsigned long long s0, unsigned long long s1) {
    u32x4 g0 = tdm_g0(lds_addr, (unsigned long long)(size_t)gptr);
    i32x8 g1 = tdm_g1(dsize, pi, pa, td0, td1, tile0, tile1, tile2, s0, s1);
    i32x4 g2; g2[0] = (int)td2; g2[1] = 0; g2[2] = 0; g2[3] = 0;
    i32x4 g3 = (i32x4)0;
    TDM_CALL(g0, g1, g2, g3);
}
#endif // HAVE_TDM

// fp32 -> (hi bf16, lo bf16) split of 16 values (bf16x3 scheme)
__device__ __forceinline__ void cvt16(const float* f, FragBF& hi, FragBF& lo) {
    #pragma unroll
    for (int j = 0; j < 16; ++j) {
        const __bf16 h = (__bf16)f[j];
        hi.e[j] = h;
        lo.e[j] = (__bf16)(f[j] - (float)h);
    }
}

#define WMMA_BF16(a, b, c) \
    __builtin_amdgcn_wmma_f32_16x16x32_bf16(false, a, false, b, (short)0, c, false, false)

// -------- Stage 1: supportT(hi/lo bf16) = (features @ weight)^T --------------
__global__ __launch_bounds__(256)
void gcn_stage1(const float* __restrict__ A, const float* __restrict__ Bw,
                __bf16* __restrict__ SThi, __bf16* __restrict__ STlo,
                int M, int N, int K)
{
    __shared__ __attribute__((aligned(16))) float sA[2][M_TILE][KPAD_F32];  // [m][k]
    __shared__ __attribute__((aligned(16))) float sB[2][N_TILE][KPAD_F32];  // [n][k]

    const int tid = threadIdx.x, wid = tid >> 5, lane = tid & 31;
    const int lhalf = lane >> 4, lm = lane & 15;
    const int wm = wid & 3, wn = wid >> 2;
    const int m0 = blockIdx.y * M_TILE, n0 = blockIdx.x * N_TILE, b = blockIdx.z;

    const float* Ab = A + (long long)b * M * K;
    const int nk = K / K_TILE;

    const v8f vzero = {0.f,0.f,0.f,0.f,0.f,0.f,0.f,0.f};
    v8f acc[2][4];
    #pragma unroll
    for (int t = 0; t < 2; ++t)
        #pragma unroll
        for (int s = 0; s < 4; ++s) acc[t][s] = vzero;

#if HAVE_TDM
    const unsigned ldsA[2] = {(unsigned)(size_t)&sA[0][0][0], (unsigned)(size_t)&sA[1][0][0]};
    const unsigned ldsB[2] = {(unsigned)(size_t)&sB[0][0][0], (unsigned)(size_t)&sB[1][0][0]};
#endif

    auto issue_tiles = [&](int kt, int buf) {
        const int k0 = kt * K_TILE;
#if HAVE_TDM
        if (wid == 0) {
            // A: 2D fp32, x walks 32 k (contiguous), y walks 128 rows stride K
            tdm_issue(ldsA[buf], Ab + (long long)m0 * K + k0, 2, 4, 3,
                      (unsigned)K, (unsigned)M, 1u << 20,
                      K_TILE, M_TILE, 0, (unsigned long long)K, 0ull);
            // weight: 3D fp32 transposing gather -> LDS [n][k]
            tdm_issue(ldsB[buf], Bw + (long long)k0 * N + n0, 2, 4, 3,
                      (unsigned)N, 1u << 20, 1u << 20,
                      1, K_TILE, N_TILE, (unsigned long long)N, 1ull);
        }
#else
        #pragma unroll
        for (int i = 0; i < 4; ++i) {
            const int linear = tid + 256 * i;
            const int m = linear >> 3, kg = linear & 7;
            const float4 f = *reinterpret_cast<const float4*>(Ab + (long long)(m0 + m) * K + k0 + 4 * kg);
            *reinterpret_cast<float4*>(&sA[buf][m][4 * kg]) = f;
        }
        #pragma unroll
        for (int i = 0; i < 4; ++i) {
            const int linear = tid + 256 * i;
            const int k = linear >> 5, ng = linear & 31;
            const float4 f = *reinterpret_cast<const float4*>(Bw + (long long)(k0 + k) * N + n0 + 4 * ng);
            sB[buf][4 * ng + 0][k] = f.x;
            sB[buf][4 * ng + 1][k] = f.y;
            sB[buf][4 * ng + 2][k] = f.z;
            sB[buf][4 * ng + 3][k] = f.w;
        }
#endif
    };
    auto wait_tiles = [&]() {
#if HAVE_TDM
        if (wid == 0) __builtin_amdgcn_s_wait_tensorcnt(0);
#endif
    };

    issue_tiles(0, 0);
    wait_tiles();
    __syncthreads();

    int buf = 0;
    for (int kt = 0; kt < nk; ++kt) {
        if (kt + 1 < nk) issue_tiles(kt + 1, buf ^ 1);

        FragBF aH[2], aL[2], bH[4], bL[4];
        #pragma unroll
        for (int t = 0; t < 2; ++t) {
            const float* p = &sA[buf][wm * 32 + t * 16 + lm][0];
            alignas(16) float f[16];
            *reinterpret_cast<float4*>(f + 0)  = *reinterpret_cast<const float4*>(p + 16 * lhalf);
            *reinterpret_cast<float4*>(f + 4)  = *reinterpret_cast<const float4*>(p + 16 * lhalf + 4);
            *reinterpret_cast<float4*>(f + 8)  = *reinterpret_cast<const float4*>(p + 16 * lhalf + 8);
            *reinterpret_cast<float4*>(f + 12) = *reinterpret_cast<const float4*>(p + 16 * lhalf + 12);
            cvt16(f, aH[t], aL[t]);
        }
        #pragma unroll
        for (int s = 0; s < 4; ++s) {
            const float* p = &sB[buf][wn * 64 + s * 16 + lm][0];
            alignas(16) float f[16];
            *reinterpret_cast<float4*>(f + 0)  = *reinterpret_cast<const float4*>(p + 16 * lhalf);
            *reinterpret_cast<float4*>(f + 4)  = *reinterpret_cast<const float4*>(p + 16 * lhalf + 4);
            *reinterpret_cast<float4*>(f + 8)  = *reinterpret_cast<const float4*>(p + 16 * lhalf + 8);
            *reinterpret_cast<float4*>(f + 12) = *reinterpret_cast<const float4*>(p + 16 * lhalf + 12);
            cvt16(f, bH[s], bL[s]);
        }

        #pragma unroll
        for (int t = 0; t < 2; ++t)
            #pragma unroll
            for (int s = 0; s < 4; ++s) {
                acc[t][s] = WMMA_BF16(aH[t].v, bH[s].v, acc[t][s]);
                acc[t][s] = WMMA_BF16(aH[t].v, bL[s].v, acc[t][s]);
                acc[t][s] = WMMA_BF16(aL[t].v, bH[s].v, acc[t][s]);
            }

        if (kt + 1 < nk) wait_tiles();
        __syncthreads();
        buf ^= 1;
    }

    // Epilogue: write supportT hi/lo bf16 planes, transposed [out][m].
    __bf16* SThi_b = SThi + (long long)b * N * M;
    __bf16* STlo_b = STlo + (long long)b * N * M;
    #pragma unroll
    for (int s = 0; s < 4; ++s) {
        const int ncol = n0 + wn * 64 + s * 16 + lm;
        #pragma unroll
        for (int t = 0; t < 2; ++t) {
            const int mbase = m0 + wm * 32 + t * 16 + 8 * lhalf;  // 8-elem aligned
            alignas(16) __bf16 ph[8], pl[8];
            #pragma unroll
            for (int r = 0; r < 8; ++r) {
                const float v = acc[t][s][r];
                const __bf16 h = (__bf16)v;
                ph[r] = h;
                pl[r] = (__bf16)(v - (float)h);
            }
            const long long off = (long long)ncol * M + mbase;
            *reinterpret_cast<uint4*>(SThi_b + off) = *reinterpret_cast<const uint4*>(ph);
            *reinterpret_cast<uint4*>(STlo_b + off) = *reinterpret_cast<const uint4*>(pl);
        }
    }
}

// -------- Stage 2: out = adjacency @ support + bias (B pre-split bf16) -------
__global__ __launch_bounds__(256)
void gcn_stage2(const float* __restrict__ A,
                const __bf16* __restrict__ Bhi, const __bf16* __restrict__ Blo,
                const float* __restrict__ bias, float* __restrict__ C,
                int M, int N, int K)
{
    __shared__ __attribute__((aligned(16))) float  sA [2][M_TILE][KPAD_F32];   // fp32 [m][k]
    __shared__ __attribute__((aligned(16))) __bf16 sBh[2][N_TILE][KPAD_BF16];  // bf16 [n][k]
    __shared__ __attribute__((aligned(16))) __bf16 sBl[2][N_TILE][KPAD_BF16];

    const int tid = threadIdx.x, wid = tid >> 5, lane = tid & 31;
    const int lhalf = lane >> 4, lm = lane & 15;
    const int wm = wid & 3, wn = wid >> 2;
    const int m0 = blockIdx.y * M_TILE, n0 = blockIdx.x * N_TILE, b = blockIdx.z;

    const float*  Ab  = A   + (long long)b * M * K;
    const __bf16* Bhb = Bhi + (long long)b * N * K;   // supportT rows: [n][k], stride K
    const __bf16* Blb = Blo + (long long)b * N * K;
    float*        Cb  = C   + (long long)b * M * N;

    const int nk = K / K_TILE;

    const v8f vzero = {0.f,0.f,0.f,0.f,0.f,0.f,0.f,0.f};
    v8f acc[2][4];
    #pragma unroll
    for (int t = 0; t < 2; ++t)
        #pragma unroll
        for (int s = 0; s < 4; ++s) acc[t][s] = vzero;

#if HAVE_TDM
    const unsigned ldsA[2]  = {(unsigned)(size_t)&sA[0][0][0],  (unsigned)(size_t)&sA[1][0][0]};
    const unsigned ldsBh[2] = {(unsigned)(size_t)&sBh[0][0][0], (unsigned)(size_t)&sBh[1][0][0]};
    const unsigned ldsBl[2] = {(unsigned)(size_t)&sBl[0][0][0], (unsigned)(size_t)&sBl[1][0][0]};
#endif

    auto issue_tiles = [&](int kt, int buf) {
        const int k0 = kt * K_TILE;
#if HAVE_TDM
        if (wid == 0) {
            // adjacency: 2D fp32, pad 4dw/32dw -> 36-dword rows
            tdm_issue(ldsA[buf], Ab + (long long)m0 * K + k0, 2, 4, 3,
                      (unsigned)K, (unsigned)M, 1u << 20,
                      K_TILE, M_TILE, 0, (unsigned long long)K, 0ull);
            // supportT planes: 2D bf16, 64B rows, pad 4dw/16dw -> 80B rows
            tdm_issue(ldsBh[buf], Bhb + (long long)n0 * K + k0, 1, 3, 3,
                      (unsigned)K, (unsigned)N, 1u << 20,
                      K_TILE, N_TILE, 0, (unsigned long long)K, 0ull);
            tdm_issue(ldsBl[buf], Blb + (long long)n0 * K + k0, 1, 3, 3,
                      (unsigned)K, (unsigned)N, 1u << 20,
                      K_TILE, N_TILE, 0, (unsigned long long)K, 0ull);
        }
#else
        #pragma unroll
        for (int i = 0; i < 4; ++i) {
            const int linear = tid + 256 * i;
            const int m = linear >> 3, kg = linear & 7;
            const float4 f = *reinterpret_cast<const float4*>(Ab + (long long)(m0 + m) * K + k0 + 4 * kg);
            *reinterpret_cast<float4*>(&sA[buf][m][4 * kg]) = f;
        }
        #pragma unroll
        for (int i = 0; i < 2; ++i) {   // hi plane: 128 cols x 32 k, 8 bf16 chunks
            const int linear = tid + 256 * i;
            const int col = linear >> 2, kg = linear & 3;
            const uint4 q = *reinterpret_cast<const uint4*>(Bhb + (long long)(n0 + col) * K + k0 + 8 * kg);
            *reinterpret_cast<uint4*>(&sBh[buf][col][8 * kg]) = q;
        }
        #pragma unroll
        for (int i = 0; i < 2; ++i) {   // lo plane
            const int linear = tid + 256 * i;
            const int col = linear >> 2, kg = linear & 3;
            const uint4 q = *reinterpret_cast<const uint4*>(Blb + (long long)(n0 + col) * K + k0 + 8 * kg);
            *reinterpret_cast<uint4*>(&sBl[buf][col][8 * kg]) = q;
        }
#endif
    };
    auto wait_tiles = [&]() {
#if HAVE_TDM
        if (wid == 0) __builtin_amdgcn_s_wait_tensorcnt(0);
#endif
    };

    issue_tiles(0, 0);
    wait_tiles();
    __syncthreads();

    int buf = 0;
    for (int kt = 0; kt < nk; ++kt) {
        if (kt + 1 < nk) issue_tiles(kt + 1, buf ^ 1);

        // A frags: fp32 -> bf16 hi/lo at load (only conversion left in stage 2)
        FragBF aH[2], aL[2], bH[4], bL[4];
        #pragma unroll
        for (int t = 0; t < 2; ++t) {
            const float* p = &sA[buf][wm * 32 + t * 16 + lm][0];
            alignas(16) float f[16];
            *reinterpret_cast<float4*>(f + 0)  = *reinterpret_cast<const float4*>(p + 16 * lhalf);
            *reinterpret_cast<float4*>(f + 4)  = *reinterpret_cast<const float4*>(p + 16 * lhalf + 4);
            *reinterpret_cast<float4*>(f + 8)  = *reinterpret_cast<const float4*>(p + 16 * lhalf + 8);
            *reinterpret_cast<float4*>(f + 12) = *reinterpret_cast<const float4*>(p + 16 * lhalf + 12);
            cvt16(f, aH[t], aL[t]);
        }
        // B frags: direct bf16, zero VALU. Lane half selects K 0-15 / 16-31 of its column.
        #pragma unroll
        for (int s = 0; s < 4; ++s) {
            const int col = wn * 64 + s * 16 + lm;
            const __bf16* ph = &sBh[buf][col][0];
            const __bf16* pl = &sBl[buf][col][0];
            bH[s].q[0] = *reinterpret_cast<const uint4*>(ph + 16 * lhalf);
            bH[s].q[1] = *reinterpret_cast<const uint4*>(ph + 16 * lhalf + 8);
            bL[s].q[0] = *reinterpret_cast<const uint4*>(pl + 16 * lhalf);
            bL[s].q[1] = *reinterpret_cast<const uint4*>(pl + 16 * lhalf + 8);
        }

        #pragma unroll
        for (int t = 0; t < 2; ++t)
            #pragma unroll
            for (int s = 0; s < 4; ++s) {
                acc[t][s] = WMMA_BF16(aH[t].v, bH[s].v, acc[t][s]);
                acc[t][s] = WMMA_BF16(aH[t].v, bL[s].v, acc[t][s]);
                acc[t][s] = WMMA_BF16(aL[t].v, bH[s].v, acc[t][s]);
            }

        if (kt + 1 < nk) wait_tiles();
        __syncthreads();
        buf ^= 1;
    }

    #pragma unroll
    for (int s = 0; s < 4; ++s) {
        const int ncol = n0 + wn * 64 + s * 16 + lm;
        const float bv = bias[ncol];
        #pragma unroll
        for (int t = 0; t < 2; ++t) {
            const int mbase = m0 + wm * 32 + t * 16 + 8 * lhalf;
            #pragma unroll
            for (int r = 0; r < 8; ++r) {
                Cb[(long long)(mbase + r) * N + ncol] = acc[t][s][r] + bv;
            }
        }
    }
}

extern "C" void kernel_launch(void* const* d_in, const int* in_sizes, int n_in,
                              void* d_out, int out_size, void* d_ws, size_t ws_size,
                              hipStream_t stream) {
    const float* adjacency = (const float*)d_in[0];  // [32,1024,1024]
    const float* features  = (const float*)d_in[1];  // [32,1024,256]
    const float* weight    = (const float*)d_in[2];  // [256,256]
    const float* bias      = (const float*)d_in[3];  // [256]
    float* out = (float*)d_out;                      // [32,1024,256]

    const int Bn = 32, Nn = 1024, INF = 256, OUTF = 256;
    __bf16* SThi = (__bf16*)d_ws;                          // [32,256,1024] bf16 (16 MB)
    __bf16* STlo = SThi + (long long)Bn * OUTF * Nn;       // [32,256,1024] bf16 (16 MB)

    dim3 block(256);
    dim3 grid(OUTF / N_TILE, Nn / M_TILE, Bn);

    // supportT(hi/lo) = (features @ weight)^T
    gcn_stage1<<<grid, block, 0, stream>>>(features, weight, SThi, STlo, Nn, OUTF, INF);
    // out = adjacency @ support + bias
    gcn_stage2<<<grid, block, 0, stream>>>(adjacency, SThi, STlo, bias, out, Nn, OUTF, Nn);
}
